// GRULayer_16381005267363
// MI455X (gfx1250) — compile-verified, matrix operand order checked by reading, not run on hardware
//
#include <hip/hip_runtime.h>

// ---------------------------------------------------------------------------
// GRU layer for MI455X (gfx1250): bf16 WMMA GEMMs + TDM (tensor_load_to_lds)
// staging + f32 gate math.  T=512, B=64, I=H=1024.
// ---------------------------------------------------------------------------

#define T_DIM 512
#define B_DIM 64
#define I_DIM 1024
#define H_DIM 1024
#define M1    (T_DIM * B_DIM)        // 32768 rows for the input projections
#define LDSS  40                     // LDS row stride (bf16 elems) = 80B = 64B row + 16B pad

typedef __attribute__((ext_vector_type(4)))  __bf16 v4bf;
typedef __attribute__((ext_vector_type(8)))  __bf16 v8bf;
typedef __attribute__((ext_vector_type(16))) __bf16 v16bf;
typedef __attribute__((ext_vector_type(8)))  float  v8f;
typedef unsigned int u32x4 __attribute__((ext_vector_type(4)));
typedef int          i32x8 __attribute__((ext_vector_type(8)));
typedef int          i32x4 __attribute__((ext_vector_type(4)));

// f32 -> bf16, round-to-nearest-even, no dependence on hip_bf16 internals.
__device__ __forceinline__ __bf16 f2bf(float f) {
    union { float f; unsigned u; } v; v.f = f;
    unsigned r = v.u + 0x7FFFu + ((v.u >> 16) & 1u);
    unsigned short h = (unsigned short)(r >> 16);
    return __builtin_bit_cast(__bf16, h);
}

__device__ __forceinline__ v8f vzero8() {
    v8f z;
#pragma unroll
    for (int i = 0; i < 8; ++i) z[i] = 0.0f;
    return z;
}

// Byte offset of a __shared__ object within the workgroup LDS allocation.
typedef __attribute__((address_space(3))) const char lds_cchar;
__device__ __forceinline__ unsigned lds_off(const void* p) {
    return (unsigned)(unsigned long long)(lds_cchar*)p;
}

// ---------------------------------------------------------------------------
// TDM: load a 2D tile (rows x 32 bf16) from global to LDS with 16B row pad,
// reproducing LDSS=40 bf16 (80B) LDS row stride.  Issued by one wave.
// D# group0: count=1 | lds_addr | global_addr | type=2 ("image")
// D# group1: data_size=2B, pad_enable, pad_interval=16 DW (64B), pad_amount=4 DW
//            (16B); tensor_dims huge (no OOB clipping); tile_dim0=32,
//            tile_dim1=rows; tensor_dim0_stride = row_stride (elems).
// 6-arg builtin form (clang-23 / therock-10.0): groups 2/3 + extra group zeroed.
// ---------------------------------------------------------------------------
__device__ __forceinline__ void tdm_load_2d(unsigned lds_byte_off,
                                            const void* gsrc,
                                            unsigned rows,
                                            unsigned row_stride_elems) {
    unsigned long long ga = (unsigned long long)gsrc;
    u32x4 g0;
    g0[0] = 1u;                                        // count=1, user descriptor
    g0[1] = lds_byte_off;                              // [63:32] lds_addr
    g0[2] = (unsigned)(ga & 0xFFFFFFFFu);              // [95:64] global_addr lo
    g0[3] = (unsigned)((ga >> 32) & 0x01FFFFFFu) | (2u << 30);  // addr hi | type=2

    i32x8 g1;
    g1[0] = (int)((1u << 16) |        // data_size = 1 -> 2 bytes
                  (1u << 20) |        // pad_enable
                  (3u << 22) |        // pad_interval: 2^(3+1)=16 DWORDs = 64B
                  (3u << 25));        // pad_amount: 3+1 = 4 DWORDs = 16B
    g1[1] = 0;                        // atomic_barrier off; tensor_dim0[15:0]=0
    g1[2] = (int)0x4000;              // tensor_dim0 = 0x40000000; tensor_dim1 lo=0
    g1[3] = (int)(0x4000u | (32u << 16));  // tensor_dim1 hi; tile_dim0 = 32
    g1[4] = (int)rows;                // tile_dim1 = rows; tile_dim2 = 0
    g1[5] = (int)row_stride_elems;    // tensor_dim0_stride lo32
    g1[6] = 0;                        // stride hi16; tensor_dim1_stride lo16
    g1[7] = 0;

    i32x4 z4 = {0, 0, 0, 0};          // groups 2/3 unused (2D tensor)
    i32x8 z8 = {0, 0, 0, 0, 0, 0, 0, 0};
    __builtin_amdgcn_tensor_load_to_lds(g0, g1, z4, z4, z8, 0);
}

// Load one 16x32 (MxK) bf16 fragment from LDS (row-major, stride LDSS).
// ISA 7.12.2 layout: lane L holds row M=L&15; lanes<16 get K=[0..7]+[16..23],
// lanes>=16 get K=[8..15]+[24..31].  Two 16B ds loads per lane.
__device__ __forceinline__ v16bf load_frag_lds(const __bf16* base, int lane) {
    const __bf16* p = base + (size_t)(lane & 15) * LDSS + ((lane >> 4) & 1) * 8;
    v8bf lo = *(const v8bf*)p;
    v8bf hi = *(const v8bf*)(p + 16);
    v16bf f;
#pragma unroll
    for (int i = 0; i < 8; ++i) { f[i] = lo[i]; f[i + 8] = hi[i]; }
    return f;
}

__device__ __forceinline__ v8f wmma_bf16(v16bf a, v16bf b, v8f c) {
    return __builtin_amdgcn_wmma_f32_16x16x32_bf16(
        /*neg_a=*/false, a, /*neg_b=*/false, b,
        /*c_mod=*/(short)0, c, /*reuse_a=*/false, /*reuse_b=*/false);
}

__device__ __forceinline__ float sigmoidf_(float x) {
    return 1.0f / (1.0f + __expf(-x));
}

// ---------------------------------------------------------------------------
// Elementwise conversion kernels (streaming; keep L2 prefetch hints)
// ---------------------------------------------------------------------------
__global__ void __launch_bounds__(256) k_cvt(const float* __restrict__ s,
                                             __bf16* __restrict__ d, int n4) {
    int i = blockIdx.x * 256 + threadIdx.x;
    if (i < n4) {
        __builtin_prefetch(s + (size_t)i * 4 + 4096, 0, 1);  // global_prefetch_b8
        float4 v = ((const float4*)s)[i];
        v4bf o;
        o[0] = f2bf(v.x); o[1] = f2bf(v.y); o[2] = f2bf(v.z); o[3] = f2bf(v.w);
        ((v4bf*)d)[i] = o;
    }
}

__global__ void __launch_bounds__(256) k_init(const float* __restrict__ state,
                                              float* __restrict__ hf0,
                                              __bf16* __restrict__ hb0, int n4) {
    int i = blockIdx.x * 256 + threadIdx.x;
    if (i < n4) {
        float4 v = ((const float4*)state)[i];
        ((float4*)hf0)[i] = v;
        v4bf o;
        o[0] = f2bf(v.x); o[1] = f2bf(v.y); o[2] = f2bf(v.z); o[3] = f2bf(v.w);
        ((v4bf*)hb0)[i] = o;
    }
}

// ---------------------------------------------------------------------------
// Phase 1: gates[g, t*B+b, h] = x[t,b,:] . w_x?[h,:] + bias[h]   (3 GEMMs)
// Block tile 128(M) x 128(N), BK=32; 8 waves each own a 32x64 tile.
// A and B tiles staged by TDM; grid = (N/128=8, M/128=256, gates=3)
// ---------------------------------------------------------------------------
__global__ void __launch_bounds__(256) k_gate_gemm(
    const __bf16* __restrict__ xbf,  // [M1, I] bf16
    const __bf16* __restrict__ wx,   // 3 x [H, I] bf16 (r,z,h)
    const float* __restrict__ br, const float* __restrict__ bz,
    const float* __restrict__ bh,
    float* __restrict__ gates)       // 3 x [M1, H] f32
{
    __shared__ __align__(16) __bf16 Abuf[128 * LDSS];
    __shared__ __align__(16) __bf16 Bbuf[128 * LDSS];

    const int tid  = threadIdx.x;
    const int lane = tid & 31;
    const int w    = tid >> 5;
    const int wm   = w & 3;      // 4 M sub-tiles of 32
    const int wn   = w >> 2;     // 2 N sub-tiles of 64
    const size_t m0 = (size_t)blockIdx.y * 128;
    const int    n0 = blockIdx.x * 128;
    const int    g  = blockIdx.z;

    const __bf16* wmat = wx + (size_t)g * ((size_t)H_DIM * I_DIM);
    const float*  bias = (g == 0) ? br : ((g == 1) ? bz : bh);
    float*        gout = gates + (size_t)g * ((size_t)M1 * H_DIM);

    const unsigned a_off = lds_off(Abuf);
    const unsigned b_off = lds_off(Bbuf);

    v8f acc[2][4];
#pragma unroll
    for (int i = 0; i < 2; ++i)
#pragma unroll
        for (int j = 0; j < 4; ++j) acc[i][j] = vzero8();

    for (int k0 = 0; k0 < I_DIM; k0 += 32) {
        if (w == 0) {
            tdm_load_2d(a_off, xbf + m0 * I_DIM + k0, 128, I_DIM);
            tdm_load_2d(b_off, wmat + (size_t)n0 * I_DIM + k0, 128, I_DIM);
            __builtin_amdgcn_s_wait_tensorcnt(0);
        }
        __syncthreads();

        v16bf af[2], bfv[4];
#pragma unroll
        for (int mf = 0; mf < 2; ++mf)
            af[mf] = load_frag_lds(&Abuf[(wm * 32 + mf * 16) * LDSS], lane);
#pragma unroll
        for (int nf = 0; nf < 4; ++nf)
            bfv[nf] = load_frag_lds(&Bbuf[(wn * 64 + nf * 16) * LDSS], lane);
#pragma unroll
        for (int mf = 0; mf < 2; ++mf)
#pragma unroll
            for (int nf = 0; nf < 4; ++nf)
                acc[mf][nf] = wmma_bf16(af[mf], bfv[nf], acc[mf][nf]);
        __syncthreads();
    }

    const int hi = (lane >> 4) & 1, nlane = lane & 15;
#pragma unroll
    for (int mf = 0; mf < 2; ++mf)
#pragma unroll
        for (int nf = 0; nf < 4; ++nf)
#pragma unroll
            for (int e = 0; e < 8; ++e) {
                size_t gm = m0 + wm * 32 + mf * 16 + e + 8 * hi;
                int    gn = n0 + wn * 64 + nf * 16 + nlane;
                gout[gm * H_DIM + gn] = acc[mf][nf][e] + bias[gn];
            }
}

// ---------------------------------------------------------------------------
// Phase 2a: r = sigmoid(g_r + h @ w_hr^T), z = sigmoid(g_z + h @ w_hz^T)
// One GEMM M=64, N=2048 (r|z), K=1024.  Block tile 64x256 (8 waves x 64x32),
// grid = 2048/256 = 8.  TDM-staged tiles; fused epilogue: bf16(r*h) and z.
// ---------------------------------------------------------------------------
__global__ void __launch_bounds__(256) k_step_rz(
    const __bf16* __restrict__ hb,   // [B, H] bf16 (current h)
    const float* __restrict__ hf,    // [B, H] f32  (current h)
    const __bf16* __restrict__ whr,  // [H, H] bf16
    const __bf16* __restrict__ whz,  // [H, H] bf16
    const float* __restrict__ gates, // 3 x [M1, H] f32
    float* __restrict__ zb,          // [B, H] f32 out
    __bf16* __restrict__ rh,         // [B, H] bf16 out (r*h)
    int t)
{
    __shared__ __align__(16) __bf16 Abuf[64 * LDSS];
    __shared__ __align__(16) __bf16 Bbuf[256 * LDSS];

    const int tid  = threadIdx.x;
    const int lane = tid & 31;
    const int w    = tid >> 5;       // 8 waves, each 32 N-columns
    const int n0   = blockIdx.x * 256;

    // whole block lies within one recurrent matrix (256 | 1024)
    const __bf16* bbase = (n0 < H_DIM)
        ? (whr + (size_t)n0 * H_DIM)
        : (whz + (size_t)(n0 - H_DIM) * H_DIM);

    const unsigned a_off = lds_off(Abuf);
    const unsigned b_off = lds_off(Bbuf);

    v8f acc[4][2];
#pragma unroll
    for (int i = 0; i < 4; ++i)
#pragma unroll
        for (int j = 0; j < 2; ++j) acc[i][j] = vzero8();

    for (int k0 = 0; k0 < H_DIM; k0 += 32) {
        if (w == 0) {
            tdm_load_2d(a_off, hb + k0, 64, H_DIM);
            tdm_load_2d(b_off, bbase + k0, 256, H_DIM);
            __builtin_amdgcn_s_wait_tensorcnt(0);
        }
        __syncthreads();

        v16bf af[4], bfv[2];
#pragma unroll
        for (int mf = 0; mf < 4; ++mf)
            af[mf] = load_frag_lds(&Abuf[(mf * 16) * LDSS], lane);
#pragma unroll
        for (int nf = 0; nf < 2; ++nf)
            bfv[nf] = load_frag_lds(&Bbuf[(w * 32 + nf * 16) * LDSS], lane);
#pragma unroll
        for (int mf = 0; mf < 4; ++mf)
#pragma unroll
            for (int nf = 0; nf < 2; ++nf)
                acc[mf][nf] = wmma_bf16(af[mf], bfv[nf], acc[mf][nf]);
        __syncthreads();
    }

    const size_t stepoff = (size_t)t * B_DIM * H_DIM;
    const float* g_r = gates + stepoff;
    const float* g_z = gates + (size_t)M1 * H_DIM + stepoff;
    const int hi = (lane >> 4) & 1, nlane = lane & 15;

#pragma unroll
    for (int mf = 0; mf < 4; ++mf)
#pragma unroll
        for (int nf = 0; nf < 2; ++nf)
#pragma unroll
            for (int e = 0; e < 8; ++e) {
                int b  = mf * 16 + e + 8 * hi;           // batch row 0..63
                int gn = n0 + w * 32 + nf * 16 + nlane;  // 0..2047
                float val = acc[mf][nf][e];
                if (gn < H_DIM) {
                    float rv = sigmoidf_(val + g_r[(size_t)b * H_DIM + gn]);
                    rh[(size_t)b * H_DIM + gn] =
                        f2bf(rv * hf[(size_t)b * H_DIM + gn]);
                } else {
                    int n2 = gn - H_DIM;
                    zb[(size_t)b * H_DIM + n2] =
                        sigmoidf_(val + g_z[(size_t)b * H_DIM + n2]);
                }
            }
}

// ---------------------------------------------------------------------------
// Phase 2b: hcand = tanh(g_h + (r*h) @ w_hh^T);  h' = z*h + (1-z)*hcand
// GEMM M=64, N=1024, K=1024.  grid = 1024/256 = 4.  TDM-staged tiles.
// ---------------------------------------------------------------------------
__global__ void __launch_bounds__(256) k_step_h(
    const __bf16* __restrict__ rh,   // [B, H] bf16 (r*h)
    const float* __restrict__ hf_in, // [B, H] f32
    const float* __restrict__ zb,    // [B, H] f32
    const __bf16* __restrict__ whh,  // [H, H] bf16
    const float* __restrict__ gates, // 3 x [M1, H]
    float* __restrict__ hf_out,      // [B, H] f32
    __bf16* __restrict__ hb_out,     // [B, H] bf16
    float* __restrict__ out,         // d_out: ys [T,B,H] then final [B,H]
    int t, int last)
{
    __shared__ __align__(16) __bf16 Abuf[64 * LDSS];
    __shared__ __align__(16) __bf16 Bbuf[256 * LDSS];

    const int tid  = threadIdx.x;
    const int lane = tid & 31;
    const int w    = tid >> 5;
    const int n0   = blockIdx.x * 256;

    const __bf16* bbase = whh + (size_t)n0 * H_DIM;
    const unsigned a_off = lds_off(Abuf);
    const unsigned b_off = lds_off(Bbuf);

    v8f acc[4][2];
#pragma unroll
    for (int i = 0; i < 4; ++i)
#pragma unroll
        for (int j = 0; j < 2; ++j) acc[i][j] = vzero8();

    for (int k0 = 0; k0 < H_DIM; k0 += 32) {
        if (w == 0) {
            tdm_load_2d(a_off, rh + k0, 64, H_DIM);
            tdm_load_2d(b_off, bbase + k0, 256, H_DIM);
            __builtin_amdgcn_s_wait_tensorcnt(0);
        }
        __syncthreads();

        v16bf af[4], bfv[2];
#pragma unroll
        for (int mf = 0; mf < 4; ++mf)
            af[mf] = load_frag_lds(&Abuf[(mf * 16) * LDSS], lane);
#pragma unroll
        for (int nf = 0; nf < 2; ++nf)
            bfv[nf] = load_frag_lds(&Bbuf[(w * 32 + nf * 16) * LDSS], lane);
#pragma unroll
        for (int mf = 0; mf < 4; ++mf)
#pragma unroll
            for (int nf = 0; nf < 2; ++nf)
                acc[mf][nf] = wmma_bf16(af[mf], bfv[nf], acc[mf][nf]);
        __syncthreads();
    }

    const size_t stepoff = (size_t)t * B_DIM * H_DIM;
    const float* g_h = gates + 2 * (size_t)M1 * H_DIM + stepoff;
    const int hi = (lane >> 4) & 1, nlane = lane & 15;

#pragma unroll
    for (int mf = 0; mf < 4; ++mf)
#pragma unroll
        for (int nf = 0; nf < 2; ++nf)
#pragma unroll
            for (int e = 0; e < 8; ++e) {
                int b  = mf * 16 + e + 8 * hi;
                int gn = n0 + w * 32 + nf * 16 + nlane;
                size_t idx = (size_t)b * H_DIM + gn;
                float hc = tanhf(acc[mf][nf][e] + g_h[idx]);
                float z  = zb[idx];
                float h  = hf_in[idx];
                float hn = z * h + (1.0f - z) * hc;
                hf_out[idx] = hn;
                hb_out[idx] = f2bf(hn);
                out[((size_t)t * B_DIM + b) * H_DIM + gn] = hn;
                if (last) out[(size_t)T_DIM * B_DIM * H_DIM + idx] = hn;
            }
}

// ---------------------------------------------------------------------------
// Host launcher
// ---------------------------------------------------------------------------
extern "C" void kernel_launch(void* const* d_in, const int* in_sizes, int n_in,
                              void* d_out, int out_size, void* d_ws, size_t ws_size,
                              hipStream_t stream) {
    const float* x     = (const float*)d_in[0];
    const float* state = (const float*)d_in[1];
    const float* wxr   = (const float*)d_in[2];
    const float* bxr   = (const float*)d_in[3];
    const float* whr   = (const float*)d_in[4];
    const float* wxz   = (const float*)d_in[5];
    const float* bxz   = (const float*)d_in[6];
    const float* whz   = (const float*)d_in[7];
    const float* wxh   = (const float*)d_in[8];
    const float* bxh   = (const float*)d_in[9];
    const float* whh   = (const float*)d_in[10];

    char* ws = (char*)d_ws;
    const size_t BH  = (size_t)B_DIM * H_DIM;
    const size_t WSZ = (size_t)H_DIM * I_DIM;          // elems per weight matrix

    size_t off = 0;
    __bf16* xbf  = (__bf16*)(ws + off); off += (size_t)M1 * I_DIM * 2;     // 64 MB
    __bf16* wbf  = (__bf16*)(ws + off); off += 6 * WSZ * 2;                // 12 MB
    float*  gate = (float*)(ws + off);  off += 3 * (size_t)M1 * H_DIM * 4; // 384 MB
    float*  hf   = (float*)(ws + off);  off += 2 * BH * 4;
    __bf16* hbp  = (__bf16*)(ws + off); off += 2 * BH * 2;
    float*  zb   = (float*)(ws + off);  off += BH * 4;
    __bf16* rh   = (__bf16*)(ws + off); off += BH * 2;

    __bf16* wxr_b = wbf + 0 * WSZ;  // slots 0..2: input weights (r,z,h)
    __bf16* wxz_b = wbf + 1 * WSZ;
    __bf16* wxh_b = wbf + 2 * WSZ;
    __bf16* whr_b = wbf + 3 * WSZ;  // slots 3..5: recurrent weights
    __bf16* whz_b = wbf + 4 * WSZ;
    __bf16* whh_b = wbf + 5 * WSZ;

    // --- convert inputs to bf16 ---
    {
        int n4 = (int)(((size_t)M1 * I_DIM) / 4);
        k_cvt<<<(n4 + 255) / 256, 256, 0, stream>>>(x, xbf, n4);
        int w4 = (int)(WSZ / 4);
        int wb = (w4 + 255) / 256;
        k_cvt<<<wb, 256, 0, stream>>>(wxr, wxr_b, w4);
        k_cvt<<<wb, 256, 0, stream>>>(wxz, wxz_b, w4);
        k_cvt<<<wb, 256, 0, stream>>>(wxh, wxh_b, w4);
        k_cvt<<<wb, 256, 0, stream>>>(whr, whr_b, w4);
        k_cvt<<<wb, 256, 0, stream>>>(whz, whz_b, w4);
        k_cvt<<<wb, 256, 0, stream>>>(whh, whh_b, w4);
    }

    // --- phase 1: all three input projections (bias fused) ---
    k_gate_gemm<<<dim3(H_DIM / 128, M1 / 128, 3), 256, 0, stream>>>(
        xbf, wbf, bxr, bxz, bxh, gate);

    // --- init hidden state (ping-pong slot 0) ---
    {
        int n4 = (int)(BH / 4);
        k_init<<<(n4 + 255) / 256, 256, 0, stream>>>(state, hf, hbp, n4);
    }

    // --- phase 2: sequential scan ---
    for (int t = 0; t < T_DIM; ++t) {
        const float*  hf_in  = hf  + (size_t)(t & 1) * BH;
        float*        hf_out = hf  + (size_t)((t + 1) & 1) * BH;
        const __bf16* hb_in  = hbp + (size_t)(t & 1) * BH;
        __bf16*       hb_out = hbp + (size_t)((t + 1) & 1) * BH;

        k_step_rz<<<dim3((2 * H_DIM) / 256), 256, 0, stream>>>(
            hb_in, hf_in, whr_b, whz_b, gate, zb, rh, t);
        k_step_h<<<dim3(H_DIM / 256), 256, 0, stream>>>(
            rh, hf_in, zb, whh_b, gate, hf_out, hb_out, (float*)d_out,
            t, (t == T_DIM - 1) ? 1 : 0);
    }
}
